// GCN_609885356937
// MI455X (gfx1250) — compile-verified, matrix operand order checked by reading, not run on hardware
//
#include <hip/hip_runtime.h>

#define N_NODES    100000
#define N_EDGES    400000
#define IN_DIM     69
#define KPAD1      96            // IN_DIM padded to multiple of 32
#define HIDDEN     256
#define NUM_CLASSES 2
#define NUM_GRAPHS 4096
#define ROW_PAD    128           // slack rows so edge-tile async loads never fault

typedef __attribute__((ext_vector_type(16))) _Float16 v16h;
typedef __attribute__((ext_vector_type(8)))  _Float16 v8h;
typedef __attribute__((ext_vector_type(8)))  float    v8f;

// ---------------------------------------------------------------------------
// CDNA5 async global->LDS helpers (ASYNCcnt path, ISA 08_async_tensor §4)
// ---------------------------------------------------------------------------
__device__ __forceinline__ uint32_t lds_offset(const void* p) {
    // generic LDS pointer = {SHARED_BASE[63:32], lds_byte_offset[31:0]}
    return (uint32_t)(uintptr_t)p;
}
__device__ __forceinline__ void async_copy_b128(uint32_t lds_off, const void* gptr) {
    asm volatile("global_load_async_to_lds_b128 %0, %1, off"
                 :: "v"(lds_off), "v"((uint64_t)(uintptr_t)gptr)
                 : "memory");
}
__device__ __forceinline__ void wait_async0() {
    asm volatile("s_wait_asynccnt 0" ::: "memory");
}

// ---------------------------------------------------------------------------
// degree / normalization
// ---------------------------------------------------------------------------
__global__ void k_deg_init(float* __restrict__ deg) {
    int i = blockIdx.x * blockDim.x + threadIdx.x;
    if (i < N_NODES) deg[i] = 1.0f;            // self-loop
}

__global__ void k_deg_edges(float* __restrict__ deg, const int* __restrict__ ei) {
    int e = blockIdx.x * blockDim.x + threadIdx.x;
    if (e < N_EDGES) atomicAdd(&deg[ei[N_EDGES + e]], 1.0f);
}

__global__ void k_deg_to_dinv(float* __restrict__ deg) {
    int i = blockIdx.x * blockDim.x + threadIdx.x;
    if (i < N_NODES) deg[i] = rsqrtf(deg[i]);  // deg >= 1 always
}

// ---------------------------------------------------------------------------
// f32 -> f16 conversion kernels (pad K with zeros; pre-transpose weights)
// ---------------------------------------------------------------------------
__global__ void k_convert_x(const float* __restrict__ x, _Float16* __restrict__ xh) {
    size_t t = (size_t)blockIdx.x * blockDim.x + threadIdx.x;
    if (t < (size_t)N_NODES * KPAD1) {
        int i = (int)(t / KPAD1), k = (int)(t % KPAD1);
        xh[t] = (k < IN_DIM) ? (_Float16)x[(size_t)i * IN_DIM + k] : (_Float16)0.0f;
    }
}

// W[K x 256] row-major -> WT[n=256][Kpad] f16 (zero-padded K)
__global__ void k_convert_wT(const float* __restrict__ W, _Float16* __restrict__ WT,
                             int K, int Kpad) {
    int t = blockIdx.x * blockDim.x + threadIdx.x;
    if (t < Kpad * HIDDEN) {
        int n = t / Kpad, k = t % Kpad;
        WT[t] = (k < K) ? (_Float16)W[(size_t)k * HIDDEN + n] : (_Float16)0.0f;
    }
}

__global__ void k_convert_h(const float* __restrict__ h, _Float16* __restrict__ hh) {
    size_t t = (size_t)blockIdx.x * blockDim.x + threadIdx.x;
    if (t < (size_t)N_NODES * HIDDEN) hh[t] = (_Float16)h[t];
}

// ---------------------------------------------------------------------------
// WMMA GEMM:  C[M x 256] = A[M x Kpad](f16) * BT[256 x Kpad](f16, transposed)
// 128x64 block tile, 8 waves, 32x32 per wave (2x2 wmma 16x16x32), f32 acc.
// Tiles staged via global_load_async_to_lds_b128 with double buffering.
// ---------------------------------------------------------------------------
#define BM 128
#define BN 64
#define BK 32

__global__ __launch_bounds__(256) void k_gemm_wmma(
        const _Float16* __restrict__ A, const _Float16* __restrict__ BT,
        float* __restrict__ C, int M, int Kpad)
{
    __shared__ __align__(16) _Float16 sA[2][BM][BK];   // [m][k]
    __shared__ __align__(16) _Float16 sB[2][BN][BK];   // [n][k]

    const int tid  = threadIdx.x;
    const int lane = tid & 31;
    const int wave = tid >> 5;
    const int bm   = blockIdx.x * BM;
    const int bn   = blockIdx.y * BN;
    const int wm   = (wave & 3) * 32;
    const int wn   = (wave >> 2) * 32;
    const int hl   = lane >> 4;
    const int l16  = lane & 15;

    // per-thread tile-copy coordinates (16B granules)
    const int am0 = tid >> 2,          akc0 = (tid & 3) * 8;          // A granule 0
    const int am1 = (tid + 256) >> 2,  akc1 = (tid & 3) * 8;          // A granule 1
    const int bn0 = tid >> 2,          bkc0 = (tid & 3) * 8;          // B granule

    auto stage_tiles = [&](int buf, int k0) {
        async_copy_b128(lds_offset(&sA[buf][am0][akc0]),
                        A + (size_t)(bm + am0) * Kpad + k0 + akc0);
        async_copy_b128(lds_offset(&sA[buf][am1][akc1]),
                        A + (size_t)(bm + am1) * Kpad + k0 + akc1);
        async_copy_b128(lds_offset(&sB[buf][bn0][bkc0]),
                        BT + (size_t)(bn + bn0) * Kpad + k0 + bkc0);
    };

    v8f acc[2][2] = {};

    const int nsteps = Kpad >> 5;
    stage_tiles(0, 0);

    for (int s = 0; s < nsteps; ++s) {
        const int buf = s & 1;
        wait_async0();          // my async copies for tile s are done
        __syncthreads();        // => everyone's copies are done
        if (s + 1 < nsteps) stage_tiles(buf ^ 1, (s + 1) * BK);

        v16h af[2], bf[2];
#pragma unroll
        for (int i = 0; i < 2; ++i) {
            const _Float16* base = &sA[buf][wm + i * 16 + l16][hl * 8];
            v8h lo = *(const v8h*)(base);
            v8h hi = *(const v8h*)(base + 16);
            af[i] = __builtin_shufflevector(lo, hi, 0,1,2,3,4,5,6,7,8,9,10,11,12,13,14,15);
        }
#pragma unroll
        for (int j = 0; j < 2; ++j) {
            const _Float16* base = &sB[buf][wn + j * 16 + l16][hl * 8];
            v8h lo = *(const v8h*)(base);
            v8h hi = *(const v8h*)(base + 16);
            bf[j] = __builtin_shufflevector(lo, hi, 0,1,2,3,4,5,6,7,8,9,10,11,12,13,14,15);
        }

#pragma unroll
        for (int i = 0; i < 2; ++i)
#pragma unroll
            for (int j = 0; j < 2; ++j)
                acc[i][j] = __builtin_amdgcn_wmma_f32_16x16x32_f16(
                    false, af[i], false, bf[j], (short)0, acc[i][j], false, false);
    }

    // store: C/D layout -> row = base + v + 8*hl, col = base + l16
#pragma unroll
    for (int i = 0; i < 2; ++i)
#pragma unroll
        for (int j = 0; j < 2; ++j) {
            int col = bn + wn + j * 16 + l16;
#pragma unroll
            for (int v = 0; v < 8; ++v) {
                int row = bm + wm + i * 16 + v + 8 * hl;
                if (row < M) C[(size_t)row * HIDDEN + col] = acc[i][j][v];
            }
        }
}

// ---------------------------------------------------------------------------
// aggregation: out = b + dinv^2 * h   (self loop + bias), then edge scatter
// ---------------------------------------------------------------------------
__global__ void k_agg_init(float* __restrict__ out, const float* __restrict__ h,
                           const float* __restrict__ dinv, const float* __restrict__ bias) {
    size_t t = (size_t)blockIdx.x * blockDim.x + threadIdx.x;
    if (t < (size_t)N_NODES * HIDDEN) {
        int i = (int)(t >> 8);
        int f = (int)(t & 255);
        float w = dinv[i];
        out[t] = bias[f] + h[t] * w * w;
    }
}

__global__ void k_agg_edges(float* __restrict__ out, const float* __restrict__ h,
                            const float* __restrict__ dinv, const int* __restrict__ ei) {
    size_t t = (size_t)blockIdx.x * blockDim.x + threadIdx.x;
    if (t >= (size_t)N_EDGES * 64) return;
    int e  = (int)(t >> 6);
    int f0 = (int)(t & 63) * 4;
    int s = ei[e];
    int d = ei[N_EDGES + e];
    float w = dinv[s] * dinv[d];
    const float4 hv = *(const float4*)&h[(size_t)s * HIDDEN + f0];
    float* o = &out[(size_t)d * HIDDEN + f0];
    atomicAdd(o + 0, hv.x * w);
    atomicAdd(o + 1, hv.y * w);
    atomicAdd(o + 2, hv.z * w);
    atomicAdd(o + 3, hv.w * w);
}

__global__ void k_relu(float* __restrict__ x) {
    size_t t = (size_t)blockIdx.x * blockDim.x + threadIdx.x;
    if (t < (size_t)N_NODES * HIDDEN) x[t] = fmaxf(x[t], 0.0f);
}

// ---------------------------------------------------------------------------
// mean pool + final linear
// ---------------------------------------------------------------------------
__global__ void k_pool_zero(float* __restrict__ pooled, float* __restrict__ cnt) {
    int t = blockIdx.x * blockDim.x + threadIdx.x;
    if (t < NUM_GRAPHS * HIDDEN) pooled[t] = 0.0f;
    if (t < NUM_GRAPHS) cnt[t] = 0.0f;
}

__global__ void k_pool_cnt(float* __restrict__ cnt, const int* __restrict__ batch) {
    int i = blockIdx.x * blockDim.x + threadIdx.x;
    if (i < N_NODES) atomicAdd(&cnt[batch[i]], 1.0f);
}

__global__ void k_pool_sum(float* __restrict__ pooled, const float* __restrict__ h,
                           const int* __restrict__ batch) {
    size_t t = (size_t)blockIdx.x * blockDim.x + threadIdx.x;
    if (t < (size_t)N_NODES * HIDDEN) {
        int i = (int)(t >> 8);
        int f = (int)(t & 255);
        atomicAdd(&pooled[(size_t)batch[i] * HIDDEN + f], h[t]);
    }
}

__global__ void k_final_linear(const float* __restrict__ pooled, const float* __restrict__ cnt,
                               const float* __restrict__ Wl, const float* __restrict__ bl,
                               float* __restrict__ out) {
    int t = blockIdx.x * blockDim.x + threadIdx.x;
    if (t >= NUM_GRAPHS * NUM_CLASSES) return;
    int g = t >> 1, c = t & 1;
    float inv = 1.0f / fmaxf(cnt[g], 1.0f);
    float s = 0.0f;
    const float* p = &pooled[(size_t)g * HIDDEN];
    for (int f = 0; f < HIDDEN; ++f) s += p[f] * Wl[f * NUM_CLASSES + c];
    out[t] = s * inv + bl[c];
}

// ---------------------------------------------------------------------------
// launch
// ---------------------------------------------------------------------------
extern "C" void kernel_launch(void* const* d_in, const int* in_sizes, int n_in,
                              void* d_out, int out_size, void* d_ws, size_t ws_size,
                              hipStream_t stream) {
    const float* x     = (const float*)d_in[0];
    const int*   ei    = (const int*)  d_in[1];
    const int*   batch = (const int*)  d_in[2];
    const float* W[4]  = {(const float*)d_in[3], (const float*)d_in[5],
                          (const float*)d_in[7], (const float*)d_in[9]};
    const float* b[4]  = {(const float*)d_in[4], (const float*)d_in[6],
                          (const float*)d_in[8], (const float*)d_in[10]};
    const float* Wl = (const float*)d_in[11];
    const float* bl = (const float*)d_in[12];
    float* out = (float*)d_out;

    // workspace carve-up
    char* ws = (char*)d_ws;
    size_t off = 0;
    auto carve = [&](size_t bytes) -> void* {
        void* p = ws + off;
        off += (bytes + 255) & ~(size_t)255;
        return p;
    };
    float*    dinv   = (float*)   carve((size_t)N_NODES * 4);
    float*    bufA   = (float*)   carve((size_t)N_NODES * HIDDEN * 4);
    float*    bufB   = (float*)   carve((size_t)N_NODES * HIDDEN * 4);
    float*    pooled = (float*)   carve((size_t)NUM_GRAPHS * HIDDEN * 4);
    float*    cnt    = (float*)   carve((size_t)NUM_GRAPHS * 4);
    _Float16* Xh     = (_Float16*)carve((size_t)(N_NODES + ROW_PAD) * KPAD1 * 2);
    _Float16* Hh     = (_Float16*)carve((size_t)(N_NODES + ROW_PAD) * HIDDEN * 2);
    _Float16* WTh    = (_Float16*)carve((size_t)HIDDEN * HIDDEN * 2);

    const int T = 256;
    auto nb = [](size_t n) { return (unsigned)((n + 255) / 256); };

    // normalization coefficients
    k_deg_init   <<<nb(N_NODES), T, 0, stream>>>(dinv);
    k_deg_edges  <<<nb(N_EDGES), T, 0, stream>>>(dinv, ei);
    k_deg_to_dinv<<<nb(N_NODES), T, 0, stream>>>(dinv);

    // layer-1 input in f16, K padded to 96
    k_convert_x<<<nb((size_t)N_NODES * KPAD1), T, 0, stream>>>(x, Xh);

    for (int l = 0; l < 4; ++l) {
        const _Float16* Ain;
        int K, Kpad;
        if (l == 0) { Ain = Xh; K = IN_DIM; Kpad = KPAD1; }
        else {
            k_convert_h<<<nb((size_t)N_NODES * HIDDEN), T, 0, stream>>>(bufA, Hh);
            Ain = Hh; K = HIDDEN; Kpad = HIDDEN;
        }
        k_convert_wT<<<nb((size_t)Kpad * HIDDEN), T, 0, stream>>>(W[l], WTh, K, Kpad);

        dim3 g((N_NODES + BM - 1) / BM, HIDDEN / BN);
        k_gemm_wmma<<<g, T, 0, stream>>>(Ain, WTh, bufB, N_NODES, Kpad);

        k_agg_init <<<nb((size_t)N_NODES * HIDDEN), T, 0, stream>>>(bufA, bufB, dinv, b[l]);
        k_agg_edges<<<nb((size_t)N_EDGES * 64),     T, 0, stream>>>(bufA, bufB, dinv, ei);
        if (l < 3) k_relu<<<nb((size_t)N_NODES * HIDDEN), T, 0, stream>>>(bufA);
    }

    // global mean pool + classifier
    k_pool_zero<<<nb(NUM_GRAPHS * HIDDEN), T, 0, stream>>>(pooled, cnt);
    k_pool_cnt <<<nb(N_NODES), T, 0, stream>>>(cnt, batch);
    k_pool_sum <<<nb((size_t)N_NODES * HIDDEN), T, 0, stream>>>(pooled, bufA, batch);
    k_final_linear<<<nb(NUM_GRAPHS * NUM_CLASSES), T, 0, stream>>>(pooled, cnt, Wl, bl, out);
}
